// DecoderModel_79259326480848
// MI455X (gfx1250) — compile-verified, matrix operand order checked by reading, not run on hardware
//
#include <hip/hip_runtime.h>
#include <hip/hip_bf16.h>

typedef __attribute__((ext_vector_type(16))) _Float16 v16h;
typedef __attribute__((ext_vector_type(8)))  float    v8f;

// Problem constants
#define BATCH 512
#define HID   196
#define REC   488
#define NC    35
#define TSTEPS 120

// ---------------------------------------------------------------------------
// Shuffle a row-major f32 matrix [rows, cols] (leading dim ld) into WMMA f16
// fragment order for v_wmma_f32_16x16x32_f16 (A-style 16x32 tile):
//   lanes 0-15 : row = tile_r*16 + lane,      K = {0..7, 16..23} of the k-tile
//   lanes 16-31: row = tile_r*16 + lane-16,   K = {8..15, 24..31}
//   half i of v16h: k = (i&7) + ((i&8)<<1) (+8 for hi lanes)
// dst: [ntR][ntK][32 lanes][16 halves]; OOB elements -> 0 (N/K padding).
// grid = (ntR_padded, ntK), block = 32 (one wave)
// ---------------------------------------------------------------------------
__global__ void shuffle_to_frag(const float* __restrict__ src, int rows, int cols,
                                int ld, _Float16* __restrict__ dst, int ntK) {
  const int tr = blockIdx.x, tk = blockIdx.y;
  const int lane = threadIdx.x & 31;
  const int r = tr * 16 + (lane & 15);
  const int khi = (lane >> 4) << 3;   // +8 for lanes 16..31
  v16h f;
#pragma unroll
  for (int i = 0; i < 16; ++i) {
    const int k = tk * 32 + (i & 7) + ((i & 8) << 1) + khi;
    const float v = (r < rows && k < cols) ? src[(size_t)r * ld + k] : 0.0f;
    f[i] = (_Float16)v;
  }
  *(v16h*)(dst + (((size_t)tr * ntK + tk) * 32 + lane) * 16) = f;
}

__device__ __forceinline__ void store_tile(float* __restrict__ C,
                                           const float* __restrict__ bias,
                                           v8f acc, int rowBase, int col,
                                           int N, int Npad) {
  const float bv = (col < N) ? bias[col] : 0.0f;  // store-phase only; WMMA done
  float* cp = C + (size_t)rowBase * Npad + col;
#pragma unroll
  for (int v = 0; v < 8; ++v)
    cp[(size_t)v * Npad] = acc[v] + bv;
}

// ---------------------------------------------------------------------------
// C[512, Npad] = Afrag @ Wfragᵀ (+ bias). Fully padded: launch geometry covers
// exactly Npad/16 n-tiles, so the MAC loop is straight-line (EXEC all-1s for
// every WMMA). Software-pipelined: next k-tile's A + 4 B fragments are loaded
// BEFORE the current tile's 4 WMMAs issue, so WMMA execution covers L2 load
// latency instead of stalling on s_wait_loadcnt 0 per tile.
//   Afrag: [32 m-tiles][ntK][32][16] f16
//   Wfrag: [Npad/16][ntK][32][16] f16
// grid = (Npad/16 / (4*waves), 32), block = 32*waves
// ---------------------------------------------------------------------------
__global__ void gemm_wmma(const _Float16* __restrict__ Af,
                          const _Float16* __restrict__ Wf,
                          const float* __restrict__ bias,
                          float* __restrict__ C,
                          int N, int Npad, int ntK) {
  const int lane = threadIdx.x & 31;
  const int w    = threadIdx.x >> 5;
  const int wavesPerBlk = blockDim.x >> 5;
  const int mt   = blockIdx.y;                           // m-tile 0..31
  const int nt0  = (blockIdx.x * wavesPerBlk + w) << 2;  // 4 n-tiles per wave

  v8f acc0 = {}, acc1 = {}, acc2 = {}, acc3 = {};

  const size_t laneOff    = (size_t)lane * 16;
  const size_t tileStride = (size_t)ntK * 32 * 16;       // halves per tile row
  const _Float16* aPtr = Af + (size_t)mt * tileStride + laneOff;
  const _Float16* bPtr = Wf + (size_t)nt0 * tileStride + laneOff;

  // prologue: load k-tile 0
  v16h a  = *(const v16h*)(aPtr);
  v16h b0 = *(const v16h*)(bPtr);
  v16h b1 = *(const v16h*)(bPtr + tileStride);
  v16h b2 = *(const v16h*)(bPtr + 2 * tileStride);
  v16h b3 = *(const v16h*)(bPtr + 3 * tileStride);

  for (int kt = 0; kt < ntK - 1; ++kt) {       // uniform trip count: scalar branch
    aPtr += 32 * 16;
    bPtr += 32 * 16;
    // preload next k-tile before consuming the current one
    const v16h an  = *(const v16h*)(aPtr);
    const v16h bn0 = *(const v16h*)(bPtr);
    const v16h bn1 = *(const v16h*)(bPtr + tileStride);
    const v16h bn2 = *(const v16h*)(bPtr + 2 * tileStride);
    const v16h bn3 = *(const v16h*)(bPtr + 3 * tileStride);
    __builtin_prefetch(aPtr + 32 * 16, 0, 1);  // warm k-tile kt+2 in L2
    __builtin_prefetch(bPtr + 32 * 16, 0, 1);
    acc0 = __builtin_amdgcn_wmma_f32_16x16x32_f16(false, a, false, b0, (short)0, acc0, false, false);
    acc1 = __builtin_amdgcn_wmma_f32_16x16x32_f16(false, a, false, b1, (short)0, acc1, false, false);
    acc2 = __builtin_amdgcn_wmma_f32_16x16x32_f16(false, a, false, b2, (short)0, acc2, false, false);
    acc3 = __builtin_amdgcn_wmma_f32_16x16x32_f16(false, a, false, b3, (short)0, acc3, false, false);
    a = an; b0 = bn0; b1 = bn1; b2 = bn2; b3 = bn3;
  }
  // epilogue: last k-tile
  acc0 = __builtin_amdgcn_wmma_f32_16x16x32_f16(false, a, false, b0, (short)0, acc0, false, false);
  acc1 = __builtin_amdgcn_wmma_f32_16x16x32_f16(false, a, false, b1, (short)0, acc1, false, false);
  acc2 = __builtin_amdgcn_wmma_f32_16x16x32_f16(false, a, false, b2, (short)0, acc2, false, false);
  acc3 = __builtin_amdgcn_wmma_f32_16x16x32_f16(false, a, false, b3, (short)0, acc3, false, false);

  // C/D layout: VGPR v, lane l -> M = v + 8*(l>=16), N = l&15
  const int rowBase = (mt << 4) + ((lane >> 4) << 3);
  const int nlo = lane & 15;
  store_tile(C, bias, acc0, rowBase, ((nt0 + 0) << 4) + nlo, N, Npad);
  store_tile(C, bias, acc1, rowBase, ((nt0 + 1) << 4) + nlo, N, Npad);
  store_tile(C, bias, acc2, rowBase, ((nt0 + 2) << 4) + nlo, N, Npad);
  store_tile(C, bias, acc3, rowBase, ((nt0 + 3) << 4) + nlo, N, Npad);
}

// Eval-mode BatchNorm on a padded buffer: x = g[c] * (x / sqrt(1+eps)) + beta[c]
__global__ void bn_kernel(float* __restrict__ x, const float* __restrict__ g,
                          const float* __restrict__ beta, int rows, int cols, int ld) {
  const int idx = blockIdx.x * blockDim.x + threadIdx.x;
  if (idx >= rows * cols) return;
  const int r = idx / cols, c = idx - r * cols;
  const float inv = 0.99950037468777324f; // 1/sqrt(1.001)
  float* px = x + (size_t)r * ld + c;
  *px = g[c] * (*px * inv) + beta[c];
}

// ---------------------------------------------------------------------------
// GRU gate update. xg/hg: [512, ldg] with biases already folded in by gemm.
// Writes h (f32 state, in-place) and its f16 fragment mirror for the next
// step's WMMA GEMMs (padding region of hfrag stays 0 from the initial memset).
// ---------------------------------------------------------------------------
__global__ void gru_gate_update(const float* __restrict__ xg,
                                const float* __restrict__ hg,
                                float* __restrict__ h,
                                _Float16* __restrict__ hfrag,
                                int H, int ldg, int ntK) {
  const int idx = blockIdx.x * blockDim.x + threadIdx.x;
  if (idx >= BATCH * H) return;
  const int b = idx / H, i = idx - b * H;
  const float* xr = xg + (size_t)b * ldg;
  const float* hr = hg + (size_t)b * ldg;
  const float r = 1.0f / (1.0f + __expf(-(xr[i] + hr[i])));
  const float z = 1.0f / (1.0f + __expf(-(xr[H + i] + hr[H + i])));
  const float n = tanhf(xr[2 * H + i] + r * hr[2 * H + i]);
  const float hn = (1.0f - z) * n + z * h[idx];
  h[idx] = hn;
  // scatter into fragment order
  const int tm = b >> 4, tk = i >> 5, ko = i & 31;
  const int lane = (b & 15) + (((ko >> 3) & 1) << 4);
  const int ii = (ko & 7) + ((ko & 16) ? 8 : 0);
  hfrag[(((size_t)tm * ntK + tk) * 32 + lane) * 16 + ii] = (_Float16)hn;
}

// ---------------------------------------------------------------------------
// CustomGRUCell step (NC=35, tiny): one workgroup, f32 VALU.
// xgc: [512, ldx] = x_t @ cell_Wihᵀ + cell_bih (from WMMA gemm).
// Note Python precedence: n = softmax(xn + (r*h) @ Whh_nᵀ + b_n).
// ---------------------------------------------------------------------------
__global__ void cell_step(const float* __restrict__ xgc, int ldx,
                          const float* __restrict__ Whh,   // [3*NC, NC]
                          float* __restrict__ hc,          // [512, NC] state
                          float* __restrict__ u_s,
                          float* __restrict__ rh_s,
                          float* __restrict__ np_s,
                          float* __restrict__ out, int t) {
  // Phase A: r,u gates + r*h
  for (int idx = threadIdx.x; idx < BATCH * NC; idx += blockDim.x) {
    const int b = idx / NC, i = idx - b * NC;
    const float* hb = hc + (size_t)b * NC;
    float hr = 0.f, hz = 0.f;
    for (int k = 0; k < NC; ++k) {
      const float hv = hb[k];
      hr += hv * Whh[i * NC + k];
      hz += hv * Whh[(NC + i) * NC + k];
    }
    const float r = 1.0f / (1.0f + __expf(-(xgc[(size_t)b * ldx + i] + hr)));
    const float u = 1.0f / (1.0f + __expf(-(xgc[(size_t)b * ldx + NC + i] + hz)));
    u_s[idx] = u;
    rh_s[idx] = r * hb[i];
  }
  __syncthreads();
  // Phase B: candidate pre-softmax = xn + (r*h)@Whh_nᵀ  (b_n already in xgc)
  for (int idx = threadIdx.x; idx < BATCH * NC; idx += blockDim.x) {
    const int b = idx / NC, i = idx - b * NC;
    const float* rb = rh_s + (size_t)b * NC;
    float acc = xgc[(size_t)b * ldx + 2 * NC + i];
    for (int k = 0; k < NC; ++k)
      acc += rb[k] * Whh[(2 * NC + i) * NC + k];
    np_s[idx] = acc;
  }
  __syncthreads();
  // Phase C: row softmax + state update + output
  for (int b = threadIdx.x; b < BATCH; b += blockDim.x) {
    const float* np = np_s + (size_t)b * NC;
    float mx = -1e30f;
    for (int i = 0; i < NC; ++i) mx = fmaxf(mx, np[i]);
    float s = 0.f;
    for (int i = 0; i < NC; ++i) s += __expf(np[i] - mx);
    const float invs = 1.0f / s;
    for (int i = 0; i < NC; ++i) {
      const float n = __expf(np[i] - mx) * invs;
      const float u = u_s[(size_t)b * NC + i];
      const float h2 = (1.0f - u) * n + u * hc[(size_t)b * NC + i];
      hc[(size_t)b * NC + i] = h2;
      out[((size_t)b * TSTEPS + t) * NC + i] = h2;
    }
  }
}

// ---------------------------------------------------------------------------
extern "C" void kernel_launch(void* const* d_in, const int* in_sizes, int n_in,
                              void* d_out, int out_size, void* d_ws, size_t ws_size,
                              hipStream_t stream) {
  const float* z_in   = (const float*)d_in[0];
  const float* W0     = (const float*)d_in[1];
  const float* b0     = (const float*)d_in[2];
  const float* g0     = (const float*)d_in[3];
  const float* beta0  = (const float*)d_in[4];
  const float* W1     = (const float*)d_in[5];
  const float* b1     = (const float*)d_in[6];
  const float* g1     = (const float*)d_in[7];
  const float* beta1  = (const float*)d_in[8];
  const float* Wih0   = (const float*)d_in[9];    // [1464,196]
  const float* Whh0   = (const float*)d_in[10];   // [1464,488]
  const float* bih0   = (const float*)d_in[11];
  const float* bhh0   = (const float*)d_in[12];
  const float* Wih1   = (const float*)d_in[13];   // [1464,488]
  const float* Whh1   = (const float*)d_in[14];
  const float* bih1   = (const float*)d_in[15];
  const float* bhh1   = (const float*)d_in[16];
  const float* cWih   = (const float*)d_in[17];   // [105,488]
  const float* cWhh   = (const float*)d_in[18];   // [105,35]
  const float* cbih   = (const float*)d_in[19];
  float* out = (float*)d_out;

  // k-tile counts
  const int ntK_H = (HID + 31) / 32;   // 7   (K=196)
  const int ntK_R = (REC + 31) / 32;   // 16  (K=488)
  const int N3R = 3 * REC;             // 1464

  // Padded GEMM geometries: Npad = grid.x * waves * 4 tiles * 16 cols
  // dense (N=196):  block 128 (4 waves), grid.x 1 -> 16 tiles, Npad 256
  // big   (N=1464): block 256 (8 waves), grid.x 3 -> 96 tiles, Npad 1536
  // cell  (N=105):  block  64 (2 waves), grid.x 1 ->  8 tiles, Npad 128
  const int NT_D = 16,  NP_D = 256;
  const int NT_B = 96,  NP_B = 1536;
  const int NT_C = 8,   NP_C = 128;

  // ---- workspace carve-up ----
  char* p = (char*)d_ws;
  auto alloc = [&](size_t bytes) -> void* {
    void* r = (void*)p;
    p += (bytes + 255) & ~(size_t)255;
    return r;
  };
  auto fragBytes = [](int ntR, int ntK) -> size_t {
    return (size_t)ntR * ntK * 32 * 16 * sizeof(_Float16);
  };

  _Float16* actf   = (_Float16*)alloc(fragBytes(32, ntK_H));     // [512,196] frags
  float*    tmp1   = (float*)alloc((size_t)BATCH * NP_D * 4);    // padded dense out
  _Float16* wdf    = (_Float16*)alloc(fragBytes(NT_D, ntK_H));   // W0 then W1
  _Float16* wih0f  = (_Float16*)alloc(fragBytes(NT_B, ntK_H));
  _Float16* whh0f  = (_Float16*)alloc(fragBytes(NT_B, ntK_R));
  _Float16* wih1f  = (_Float16*)alloc(fragBytes(NT_B, ntK_R));
  _Float16* whh1f  = (_Float16*)alloc(fragBytes(NT_B, ntK_R));
  _Float16* wcf    = (_Float16*)alloc(fragBytes(NT_C, ntK_R));
  float*    xg0    = (float*)alloc((size_t)BATCH * NP_B * 4);    // time-invariant
  float*    gA     = (float*)alloc((size_t)BATCH * NP_B * 4);    // hg scratch
  float*    gB     = (float*)alloc((size_t)BATCH * NP_B * 4);    // xg1 scratch
  float*    h0     = (float*)alloc((size_t)BATCH * REC * 4);
  float*    h1     = (float*)alloc((size_t)BATCH * REC * 4);
  _Float16* h0f    = (_Float16*)alloc(fragBytes(32, ntK_R));
  _Float16* h1f    = (_Float16*)alloc(fragBytes(32, ntK_R));
  float*    hc     = (float*)alloc((size_t)BATCH * NC * 4);
  float*    xgc    = (float*)alloc((size_t)BATCH * NP_C * 4);
  float*    u_s    = (float*)alloc((size_t)BATCH * NC * 4);
  float*    rh_s   = (float*)alloc((size_t)BATCH * NC * 4);
  float*    np_s   = (float*)alloc((size_t)BATCH * NC * 4);

  const dim3 blkShuf(32), blkElw(256);

  // ---- zero states (also zeroes fragment K-padding permanently) ----
  hipMemsetAsync(h0, 0, (size_t)BATCH * REC * 4, stream);
  hipMemsetAsync(h1, 0, (size_t)BATCH * REC * 4, stream);
  hipMemsetAsync(h0f, 0, fragBytes(32, ntK_R), stream);
  hipMemsetAsync(h1f, 0, fragBytes(32, ntK_R), stream);
  hipMemsetAsync(hc, 0, (size_t)BATCH * NC * 4, stream);

  // ---- dense front: z = BN(BN(z_in@W0ᵀ+b0)@W1ᵀ+b1) ----
  shuffle_to_frag<<<dim3(32, ntK_H), blkShuf, 0, stream>>>(z_in, BATCH, HID, HID, actf, ntK_H);
  shuffle_to_frag<<<dim3(NT_D, ntK_H), blkShuf, 0, stream>>>(W0, HID, HID, HID, wdf, ntK_H);
  gemm_wmma<<<dim3(1, 32), 128, 0, stream>>>(actf, wdf, b0, tmp1, HID, NP_D, ntK_H);
  bn_kernel<<<(BATCH * HID + 255) / 256, blkElw, 0, stream>>>(tmp1, g0, beta0, BATCH, HID, NP_D);
  shuffle_to_frag<<<dim3(32, ntK_H), blkShuf, 0, stream>>>(tmp1, BATCH, HID, NP_D, actf, ntK_H);
  shuffle_to_frag<<<dim3(NT_D, ntK_H), blkShuf, 0, stream>>>(W1, HID, HID, HID, wdf, ntK_H);
  gemm_wmma<<<dim3(1, 32), 128, 0, stream>>>(actf, wdf, b1, tmp1, HID, NP_D, ntK_H);
  bn_kernel<<<(BATCH * HID + 255) / 256, blkElw, 0, stream>>>(tmp1, g1, beta1, BATCH, HID, NP_D);

  // ---- time-invariant GRU0 input gates: xg0 = z @ Wih0ᵀ + bih0 ----
  shuffle_to_frag<<<dim3(32, ntK_H), blkShuf, 0, stream>>>(tmp1, BATCH, HID, NP_D, actf, ntK_H);
  shuffle_to_frag<<<dim3(NT_B, ntK_H), blkShuf, 0, stream>>>(Wih0, N3R, HID, HID, wih0f, ntK_H);
  gemm_wmma<<<dim3(3, 32), 256, 0, stream>>>(actf, wih0f, bih0, xg0, N3R, NP_B, ntK_H);

  // ---- shuffle recurrent weights once (L2-resident thereafter) ----
  shuffle_to_frag<<<dim3(NT_B, ntK_R), blkShuf, 0, stream>>>(Whh0, N3R, REC, REC, whh0f, ntK_R);
  shuffle_to_frag<<<dim3(NT_B, ntK_R), blkShuf, 0, stream>>>(Wih1, N3R, REC, REC, wih1f, ntK_R);
  shuffle_to_frag<<<dim3(NT_B, ntK_R), blkShuf, 0, stream>>>(Whh1, N3R, REC, REC, whh1f, ntK_R);
  shuffle_to_frag<<<dim3(NT_C, ntK_R), blkShuf, 0, stream>>>(cWih, 3 * NC, REC, REC, wcf, ntK_R);

  // ---- fused time loop: GRU0 -> GRU1 -> custom cell per step ----
  const int gateBlocks = (BATCH * REC + 255) / 256;
  for (int t = 0; t < TSTEPS; ++t) {
    // GRU0: hg0 = h0@Whh0ᵀ+bhh0 ; gate -> h0, h0f
    gemm_wmma<<<dim3(3, 32), 256, 0, stream>>>(h0f, whh0f, bhh0, gA, N3R, NP_B, ntK_R);
    gru_gate_update<<<gateBlocks, blkElw, 0, stream>>>(xg0, gA, h0, h0f, REC, NP_B, ntK_R);
    // GRU1: xg1 = h0@Wih1ᵀ+bih1 ; hg1 = h1@Whh1ᵀ+bhh1 ; gate -> h1, h1f
    gemm_wmma<<<dim3(3, 32), 256, 0, stream>>>(h0f, wih1f, bih1, gB, N3R, NP_B, ntK_R);
    gemm_wmma<<<dim3(3, 32), 256, 0, stream>>>(h1f, whh1f, bhh1, gA, N3R, NP_B, ntK_R);
    gru_gate_update<<<gateBlocks, blkElw, 0, stream>>>(gB, gA, h1, h1f, REC, NP_B, ntK_R);
    // Custom cell: xgc = h1@cWihᵀ+cbih (WMMA), then tiny f32 cell step
    gemm_wmma<<<dim3(1, 32), 64, 0, stream>>>(h1f, wcf, cbih, xgc, 3 * NC, NP_C, ntK_R);
    cell_step<<<1, 256, 0, stream>>>(xgc, NP_C, cWhh, hc, u_s, rh_s, np_s, out, t);
  }
}
